// TransformerModel_20452634263939
// MI455X (gfx1250) — compile-verified, matrix operand order checked by reading, not run on hardware
//
#include <hip/hip_runtime.h>

#define NB       16
#define FDIM     512
#define MAX_LEN  200
#define K_CELLS  1024
#define KPAD     224      // 199 normalized cols + 1 rank-1 col + zero pad to 7*32
#define EPSV     1e-5f

typedef __attribute__((ext_vector_type(16))) _Float16 v16h;
typedef __attribute__((ext_vector_type(8)))  _Float16 v8h;
typedef __attribute__((ext_vector_type(8)))  float    v8f;
typedef __attribute__((ext_vector_type(4)))  unsigned u32x4;
typedef __attribute__((ext_vector_type(8)))  int      i32x8;
typedef __attribute__((ext_vector_type(4)))  int      i32x4;

#if defined(__HIP_DEVICE_COMPILE__) && __has_builtin(__builtin_amdgcn_tensor_load_to_lds)
#define HAVE_TDM 1
#else
#define HAVE_TDM 0
#endif

// ---------------------------------------------------------------------------
// K0: per (b,f): prom_qm row (32), prom_qms scalar, folded coefficients
//     A[b,f,c] = sum_k prom_qm[b,f,k] * Wq_e[k*512+f, c]   (c = 0..32)
//     A0[b,f]  = sum_k prom_qm[b,f,k] * bq_e[k*512+f]
// ---------------------------------------------------------------------------
__global__ void k_prom_coef(const float* __restrict__ prom,
                            const float* __restrict__ Wq_p, const float* __restrict__ bq_p,
                            const float* __restrict__ Wq_e, const float* __restrict__ bq_e,
                            const float* __restrict__ Wsig, const float* __restrict__ bsig,
                            float* __restrict__ Acoef, float* __restrict__ Pq) {
  int idx = blockIdx.x * blockDim.x + threadIdx.x;
  if (idx >= NB * FDIM) return;
  int b = idx / FDIM, f = idx % FDIM;

  float pr[32];
#pragma unroll
  for (int c = 0; c < 32; ++c) pr[c] = prom[b * 32 + c];

  float qm[32];
  for (int k = 0; k < 32; ++k) {
    const float* w = Wq_p + (size_t)(f * 32 + k) * 32;
    float s = bq_p[f * 32 + k];
#pragma unroll
    for (int c = 0; c < 32; ++c) s += pr[c] * w[c];
    qm[k] = s;
  }

  float ps = bsig[0];
  for (int k = 0; k < 32; ++k) ps += qm[k] * Wsig[k];
  Pq[idx] = ps;

  float a0 = 0.f;
  for (int k = 0; k < 32; ++k) a0 += qm[k] * bq_e[k * FDIM + f];

  for (int c = 0; c < 33; ++c) {
    float s = 0.f;
    for (int k = 0; k < 32; ++k) s += qm[k] * Wq_e[(size_t)(k * FDIM + f) * 33 + c];
    Acoef[(size_t)idx * 34 + c] = s;
  }
  Acoef[(size_t)idx * 34 + 33] = a0;
}

// ---------------------------------------------------------------------------
// K1: one 64-thread block per (b,f) row: QK[l] = mask*(src·A + A0), layernorm
//     over l = 1..199, write f16 row Aext[b][f][0..198], col 199 = prom_qms,
//     cols 200..223 = 0.
// ---------------------------------------------------------------------------
__global__ void k_qk_ln(const float* __restrict__ src, const float* __restrict__ mask,
                        const float* __restrict__ Acoef, const float* __restrict__ Pq,
                        _Float16* __restrict__ Aext) {
  int bf = blockIdx.x;
  int b  = bf / FDIM;
  int tid = threadIdx.x;   // 64 threads

  __shared__ float sA[34];
  __shared__ float r1[64], r2[64];
  if (tid < 34) sA[tid] = Acoef[(size_t)bf * 34 + tid];
  __syncthreads();

  float vals[4];
  float s = 0.f, s2 = 0.f;
#pragma unroll
  for (int i = 0; i < 4; ++i) {
    int lp = tid + i * 64;
    float q = 0.f;
    if (lp < MAX_LEN - 1) {
      int l = lp + 1;
      const float* sp = src + ((size_t)b * MAX_LEN + l) * 33;
      q = sA[33];
#pragma unroll
      for (int c = 0; c < 33; ++c) q += sp[c] * sA[c];
      q *= mask[b * MAX_LEN + l];
      s += q; s2 += q * q;
    }
    vals[i] = q;
  }

  r1[tid] = s; r2[tid] = s2;
  __syncthreads();
  for (int off = 32; off > 0; off >>= 1) {
    if (tid < off) { r1[tid] += r1[tid + off]; r2[tid] += r2[tid + off]; }
    __syncthreads();
  }
  const float inv_n = 1.f / (float)(MAX_LEN - 1);
  float mu  = r1[0] * inv_n;
  float var = r2[0] * inv_n - mu * mu;
  float rsg = rsqrtf(var + EPSV);

  _Float16* row = Aext + (size_t)bf * KPAD;
#pragma unroll
  for (int i = 0; i < 4; ++i) {
    int lp = tid + i * 64;
    if (lp < MAX_LEN - 1) row[lp] = (_Float16)((vals[i] - mu) * rsg);
  }
  if (tid == 0) {
    row[MAX_LEN - 1] = (_Float16)Pq[bf];          // rank-1 column (not normalized)
    for (int k = MAX_LEN; k < KPAD; ++k) row[k] = (_Float16)0.f;
  }
}

// ---------------------------------------------------------------------------
// K2: Bext[b][n][k] = atac[b][n][(k+1) % 200] (f16), zero padded K -> 224.
// ---------------------------------------------------------------------------
__global__ void k_pack_atac(const float* __restrict__ atac, _Float16* __restrict__ Bh) {
  int idx = blockIdx.x * blockDim.x + threadIdx.x;
  const int total = NB * K_CELLS * KPAD;
  if (idx >= total) return;
  int k = idx % KPAD;
  int r = idx / KPAD;
  int n = r % K_CELLS;
  int b = r / K_CELLS;
  float v = 0.f;
  if (k < MAX_LEN) {
    int l = (k + 1) % MAX_LEN;   // k=0..198 -> l=1..199 ; k=199 -> l=0 (atac_p)
    v = atac[((size_t)b * K_CELLS + n) * MAX_LEN + l];
  }
  Bh[idx] = (_Float16)v;
}

__global__ void k_f32_to_f16(const float* __restrict__ x, _Float16* __restrict__ y, int n) {
  int i = blockIdx.x * blockDim.x + threadIdx.x;
  if (i < n) y[i] = (_Float16)x[i];
}

// ---------------------------------------------------------------------------
// WMMA fragment loaders (16-bit A/B layout per ISA: lane L -> h=L>>4, r=L&15;
// two contiguous 16B chunks at K = k0+h*8 and K = k0+16+h*8).
// ---------------------------------------------------------------------------
__device__ __forceinline__ v16h load_frag(const _Float16* __restrict__ base,
                                          int row0, int k0, int stride, int lane) {
  int h = lane >> 4, r = lane & 15;
  const _Float16* p = base + (size_t)(row0 + r) * stride + k0 + h * 8;
  v8h lo = *(const v8h*)p;
  v8h hi = *(const v8h*)(p + 16);
  return __builtin_shufflevector(lo, hi, 0,1,2,3,4,5,6,7,8,9,10,11,12,13,14,15);
}

// LDS tile layout: rows of 32 f16 contiguous (stride 32).
__device__ __forceinline__ v16h lds_frag(const _Float16* base, int row0, int lane) {
  int h = lane >> 4, r = lane & 15;
  const _Float16* p = base + (row0 + r) * 32 + h * 8;
  v8h lo = *(const v8h*)p;
  v8h hi = *(const v8h*)(p + 16);
  return __builtin_shufflevector(lo, hi, 0,1,2,3,4,5,6,7,8,9,10,11,12,13,14,15);
}

__device__ __forceinline__ v8f wmma16(v16h a, v16h b, v8f c) {
  return __builtin_amdgcn_wmma_f32_16x16x32_f16(false, a, false, b, (short)0, c, false, false);
}

#if HAVE_TDM
// Issue one TDM 2-D tile load: rows x 32 f16 tile from a row-major (stride
// KPAD f16) matrix into LDS (packed rows of 32 f16). D# built per ISA 8.3/8.4.
__device__ __forceinline__ void tdm_load_tile(unsigned lds_off, const _Float16* gptr,
                                              unsigned rows) {
  unsigned long long ga = (unsigned long long)(size_t)gptr;
  u32x4 g0;
  g0[0] = 1u;                                            // count=1, user mode
  g0[1] = lds_off;                                       // lds_addr (bytes)
  g0[2] = (unsigned)(ga & 0xffffffffu);                  // global_addr[31:0]
  g0[3] = (unsigned)((ga >> 32) & 0x01ffffffu) | (2u << 30);  // addr[56:32] | type=2
  i32x8 g1;
  g1[0] = (int)(1u << 16);                               // wg_mask=0, data_size=1 (2B)
  g1[1] = (int)(((unsigned)KPAD & 0xffffu) << 16);       // tensor_dim0[15:0] @ bit48
  g1[2] = (int)((rows & 0xffffu) << 16);                 // dim0 hi=0 | tensor_dim1 lo16
  g1[3] = (int)((32u) << 16);                            // tensor_dim1 hi=0 | tile_dim0=32
  g1[4] = (int)(rows & 0xffffu);                         // tile_dim1=rows, tile_dim2=0
  g1[5] = (int)(unsigned)KPAD;                           // tensor_dim0_stride lo32
  g1[6] = 0;                                             // stride hi | dim1_stride lo
  g1[7] = 0;
  i32x4 z4 = {0, 0, 0, 0};
#if __clang_major__ >= 23
  i32x8 z8 = {0, 0, 0, 0, 0, 0, 0, 0};
  __builtin_amdgcn_tensor_load_to_lds(g0, g1, z4, z4, z8, 0);
#else
  __builtin_amdgcn_tensor_load_to_lds(g0, g1, z4, z4, 0);
#endif
}
#endif

// ---------------------------------------------------------------------------
// K3: per batch: C(512 x 1024) = Aext(512 x 224) @ Bext(224 x 1024).
// Block = 256 threads = 8 waves (4x2), block tile 128M x 64N, wave tile 32x32.
// K-step tiles (A:128x32, B:64x32 f16) staged in LDS by the Tensor Data Mover
// (double-buffered, wave 0 issues, TENSORcnt-tracked), waves feed WMMA from
// LDS. Output stored transposed (cell-major, f contiguous) f16 -> GEMM2's A.
// ---------------------------------------------------------------------------
__global__ void __launch_bounds__(256)
k_gemm1(const _Float16* __restrict__ Aext, const _Float16* __restrict__ Bh,
        _Float16* __restrict__ Ch) {
  __shared__ _Float16 smA[2][128 * 32];   // 8 KB x2
  __shared__ _Float16 smB[2][64 * 32];    // 4 KB x2

  int tid  = threadIdx.x;
  int lane = tid & 31;
  int wv   = tid >> 5;          // 0..7
  int wm   = wv >> 1;           // 0..3  (M sub-tile)
  int wn   = wv & 1;            // 0..1  (N sub-tile)

  int blk = blockIdx.x;
  int bn  = blk & 15;           // 16 N-blocks (64 cols each)
  int t   = blk >> 4;
  int bm  = t & 3;              // 4 M-blocks (128 rows each)
  int b   = t >> 2;             // batch

  const _Float16* Ag = Aext + ((size_t)b * FDIM + bm * 128) * KPAD;
  const _Float16* Bg = Bh + ((size_t)b * K_CELLS + bn * 64) * KPAD;

  v8f zero = {0.f,0.f,0.f,0.f,0.f,0.f,0.f,0.f};
  v8f acc[2][2];
#pragma unroll
  for (int i = 0; i < 2; ++i)
#pragma unroll
    for (int j = 0; j < 2; ++j) acc[i][j] = zero;

  const int NK = KPAD / 32;     // 7

#if HAVE_TDM
  if (tid < 32) {               // wave 0 drives the TDM (EXEC-independent DMA)
    tdm_load_tile((unsigned)(size_t)&smA[0][0], Ag, 128);
    tdm_load_tile((unsigned)(size_t)&smB[0][0], Bg, 64);
  }
  for (int ks = 0; ks < NK; ++ks) {
    if (tid < 32) {
      if (ks + 1 < NK) {
        int nb = (ks + 1) & 1;
        tdm_load_tile((unsigned)(size_t)&smA[nb][0], Ag + (ks + 1) * 32, 128);
        tdm_load_tile((unsigned)(size_t)&smB[nb][0], Bg + (ks + 1) * 32, 64);
        __builtin_amdgcn_s_wait_tensorcnt(2);   // current step's pair complete
      } else {
        __builtin_amdgcn_s_wait_tensorcnt(0);
      }
    }
    __syncthreads();                            // publish tile ks
    const _Float16* cA = &smA[ks & 1][0];
    const _Float16* cB = &smB[ks & 1][0];
    v16h a0 = lds_frag(cA, wm * 32,      lane);
    v16h a1 = lds_frag(cA, wm * 32 + 16, lane);
    v16h b0 = lds_frag(cB, wn * 32,      lane);
    v16h b1 = lds_frag(cB, wn * 32 + 16, lane);
    acc[0][0] = wmma16(a0, b0, acc[0][0]);
    acc[0][1] = wmma16(a0, b1, acc[0][1]);
    acc[1][0] = wmma16(a1, b0, acc[1][0]);
    acc[1][1] = wmma16(a1, b1, acc[1][1]);
    __syncthreads();                            // tile ks consumed; buffer reusable
  }
#else
  // Fallback: cooperative global->LDS staging with plain loads.
  for (int ks = 0; ks < NK; ++ks) {
    __syncthreads();
    for (int c = tid; c < 512 + 256; c += 256) {   // 16B chunks: A=512, B=256
      if (c < 512) {
        int row = c >> 2, part = c & 3;
        *(v8h*)&smA[0][row * 32 + part * 8] =
            *(const v8h*)(Ag + (size_t)row * KPAD + ks * 32 + part * 8);
      } else {
        int cc = c - 512;
        int row = cc >> 2, part = cc & 3;
        *(v8h*)&smB[0][row * 32 + part * 8] =
            *(const v8h*)(Bg + (size_t)row * KPAD + ks * 32 + part * 8);
      }
    }
    __syncthreads();
    const _Float16* cA = &smA[0][0];
    const _Float16* cB = &smB[0][0];
    v16h a0 = lds_frag(cA, wm * 32,      lane);
    v16h a1 = lds_frag(cA, wm * 32 + 16, lane);
    v16h b0 = lds_frag(cB, wn * 32,      lane);
    v16h b1 = lds_frag(cB, wn * 32 + 16, lane);
    acc[0][0] = wmma16(a0, b0, acc[0][0]);
    acc[0][1] = wmma16(a0, b1, acc[0][1]);
    acc[1][0] = wmma16(a1, b0, acc[1][0]);
    acc[1][1] = wmma16(a1, b1, acc[1][1]);
  }
#endif

  // Store transposed (cell, f) f16: lane -> N = lane&15, VGPR r -> M = h*8+r,
  // so each lane owns 8 contiguous M values => packed 16-byte store.
  int h = lane >> 4, rl = lane & 15;
  _Float16* Cb = Ch + (size_t)b * K_CELLS * FDIM;
#pragma unroll
  for (int i = 0; i < 2; ++i) {
#pragma unroll
    for (int j = 0; j < 2; ++j) {
      int n    = bn * 64 + wn * 32 + j * 16 + rl;
      int mrow = bm * 128 + wm * 32 + i * 16 + h * 8;
      v8h pk;
#pragma unroll
      for (int r = 0; r < 8; ++r) pk[r] = (_Float16)acc[i][j][r];
      *(v8h*)(Cb + (size_t)n * FDIM + mrow) = pk;
    }
  }
}

// ---------------------------------------------------------------------------
// K4: H(1024 x 128) = Ch(1024 x 512) @ W1^T, +b1, ReLU. W1 row-major (128x512)
//     is already the Bt layout. Output f16, (cell, c) row-major.
// ---------------------------------------------------------------------------
__global__ void k_gemm2(const _Float16* __restrict__ Ch, const _Float16* __restrict__ W1h,
                        const float* __restrict__ b1, _Float16* __restrict__ Hh) {
  int wave = (blockIdx.x * blockDim.x + threadIdx.x) >> 5;
  int lane = threadIdx.x & 31;
  int b   = wave >> 7;          // 32*4 tiles per batch
  int rem = wave & 127;
  int mt  = rem >> 2, nt = rem & 3;

  const _Float16* A = Ch + (size_t)b * K_CELLS * FDIM;

  v8f zero = {0.f,0.f,0.f,0.f,0.f,0.f,0.f,0.f};
  v8f acc[2][2];
#pragma unroll
  for (int i = 0; i < 2; ++i)
#pragma unroll
    for (int j = 0; j < 2; ++j) acc[i][j] = zero;

  int m0 = mt * 32, n0 = nt * 32;
  for (int ks = 0; ks < FDIM / 32; ++ks) {
    int k0 = ks * 32;
    v16h a0 = load_frag(A,   m0,      k0, FDIM, lane);
    v16h a1 = load_frag(A,   m0 + 16, k0, FDIM, lane);
    v16h b0 = load_frag(W1h, n0,      k0, FDIM, lane);
    v16h b1 = load_frag(W1h, n0 + 16, k0, FDIM, lane);
    acc[0][0] = wmma16(a0, b0, acc[0][0]);
    acc[0][1] = wmma16(a0, b1, acc[0][1]);
    acc[1][0] = wmma16(a1, b0, acc[1][0]);
    acc[1][1] = wmma16(a1, b1, acc[1][1]);
  }

  int h = lane >> 4, rl = lane & 15;
#pragma unroll
  for (int i = 0; i < 2; ++i) {
#pragma unroll
    for (int j = 0; j < 2; ++j) {
      int n  = n0 + j * 16 + rl;
      int mb = m0 + i * 16 + h * 8;
      float bias = b1[n];
#pragma unroll
      for (int r = 0; r < 8; ++r) {
        float x = acc[i][j][r] + bias;
        x = x > 0.f ? x : 0.f;
        Hh[((size_t)b * K_CELLS + mb + r) * 128 + n] = (_Float16)x;
      }
    }
  }
}

// ---------------------------------------------------------------------------
// K5: out(1024 x 64) = H(1024 x 128) @ W2^T + b2, f32 to d_out.
// ---------------------------------------------------------------------------
__global__ void k_gemm3(const _Float16* __restrict__ Hh, const _Float16* __restrict__ W2h,
                        const float* __restrict__ b2, float* __restrict__ out) {
  int wave = (blockIdx.x * blockDim.x + threadIdx.x) >> 5;
  int lane = threadIdx.x & 31;
  int b   = wave >> 6;          // 32*2 tiles per batch
  int rem = wave & 63;
  int mt  = rem >> 1, nt = rem & 1;

  const _Float16* A = Hh + (size_t)b * K_CELLS * 128;

  v8f zero = {0.f,0.f,0.f,0.f,0.f,0.f,0.f,0.f};
  v8f acc[2][2];
#pragma unroll
  for (int i = 0; i < 2; ++i)
#pragma unroll
    for (int j = 0; j < 2; ++j) acc[i][j] = zero;

  int m0 = mt * 32, n0 = nt * 32;
  for (int ks = 0; ks < 128 / 32; ++ks) {
    int k0 = ks * 32;
    v16h a0 = load_frag(A,   m0,      k0, 128, lane);
    v16h a1 = load_frag(A,   m0 + 16, k0, 128, lane);
    v16h b0 = load_frag(W2h, n0,      k0, 128, lane);
    v16h b1 = load_frag(W2h, n0 + 16, k0, 128, lane);
    acc[0][0] = wmma16(a0, b0, acc[0][0]);
    acc[0][1] = wmma16(a0, b1, acc[0][1]);
    acc[1][0] = wmma16(a1, b0, acc[1][0]);
    acc[1][1] = wmma16(a1, b1, acc[1][1]);
  }

  int h = lane >> 4, rl = lane & 15;
#pragma unroll
  for (int i = 0; i < 2; ++i) {
#pragma unroll
    for (int j = 0; j < 2; ++j) {
      int n  = n0 + j * 16 + rl;
      int mb = m0 + i * 16 + h * 8;
      float bias = b2[n];
#pragma unroll
      for (int r = 0; r < 8; ++r)
        out[((size_t)b * K_CELLS + mb + r) * 64 + n] = acc[i][j][r] + bias;
    }
  }
}

// ---------------------------------------------------------------------------
extern "C" void kernel_launch(void* const* d_in, const int* in_sizes, int n_in,
                              void* d_out, int out_size, void* d_ws, size_t ws_size,
                              hipStream_t stream) {
  (void)in_sizes; (void)n_in; (void)out_size; (void)ws_size;

  const float* src  = (const float*)d_in[0];
  const float* prom = (const float*)d_in[1];
  const float* atac = (const float*)d_in[2];
  const float* mask = (const float*)d_in[3];
  const float* Wq_p = (const float*)d_in[4];
  const float* bq_p = (const float*)d_in[5];
  const float* Wq_e = (const float*)d_in[6];
  const float* bq_e = (const float*)d_in[7];
  const float* Wsig = (const float*)d_in[8];
  const float* bsig = (const float*)d_in[9];
  const float* W1   = (const float*)d_in[10];
  const float* b1   = (const float*)d_in[11];
  const float* W2   = (const float*)d_in[12];
  const float* b2   = (const float*)d_in[13];
  float* out = (float*)d_out;

  char* p = (char*)d_ws;
  auto carve = [&](size_t bytes) -> void* {
    void* q = (void*)p;
    p += (bytes + 255) & ~(size_t)255;
    return q;
  };
  float*     Acoef = (float*)    carve((size_t)NB * FDIM * 34 * 4);
  float*     Pq    = (float*)    carve((size_t)NB * FDIM * 4);
  _Float16*  Aext  = (_Float16*) carve((size_t)NB * FDIM * KPAD * 2);
  _Float16*  Bh    = (_Float16*) carve((size_t)NB * K_CELLS * KPAD * 2);
  _Float16*  Chm   = (_Float16*) carve((size_t)NB * K_CELLS * FDIM * 2);
  _Float16*  Hh    = (_Float16*) carve((size_t)NB * K_CELLS * 128 * 2);
  _Float16*  W1h   = (_Float16*) carve((size_t)128 * FDIM * 2);
  _Float16*  W2h   = (_Float16*) carve((size_t)64 * 128 * 2);

  // Stage 0: folded prompt-side coefficients
  k_prom_coef<<<(NB * FDIM + 255) / 256, 256, 0, stream>>>(
      prom, Wq_p, bq_p, Wq_e, bq_e, Wsig, bsig, Acoef, Pq);

  // Stage 1: QK + mask + layernorm -> f16 A operand (with rank-1 column + pad)
  k_qk_ln<<<NB * FDIM, 64, 0, stream>>>(src, mask, Acoef, Pq, Aext);

  // Stage 2: pack B operand and weights to f16
  {
    int total = NB * K_CELLS * KPAD;
    k_pack_atac<<<(total + 255) / 256, 256, 0, stream>>>(atac, Bh);
  }
  k_f32_to_f16<<<(128 * FDIM + 255) / 256, 256, 0, stream>>>(W1, W1h, 128 * FDIM);
  k_f32_to_f16<<<(64 * 128 + 255) / 256, 256, 0, stream>>>(W2, W2h, 64 * 128);

  // Stage 3: big GEMM (TDM-staged, per batch 512x224 @ 224x1024)
  k_gemm1<<<NB * 4 * 16, 256, 0, stream>>>(Aext, Bh, Chm);

  // Stage 4: FC1 + bias + ReLU
  k_gemm2<<<(NB * 32 * 4) / 8, 256, 0, stream>>>(Chm, W1h, b1, Hh);

  // Stage 5: FC2 + bias -> f32 output
  k_gemm3<<<(NB * 32 * 2) / 8, 256, 0, stream>>>(Hh, W2h, b2, out);
}
